// PointNet_38981123178753
// MI455X (gfx1250) — matrix operand order repaired and checked by reading, not compiled, not run on hardware
//
#include <hip/hip_runtime.h>
#include <hip/hip_bf16.h>
#include <math.h>

typedef __attribute__((ext_vector_type(16))) _Float16 v16h;
typedef __attribute__((ext_vector_type(8)))  _Float16 v8h;
typedef __attribute__((ext_vector_type(8)))  float    v8f;

#define KNN 20
#define BN_EPS 1e-5f
#define LRELU 0.2f

// ---------------------------------------------------------------------------
// (16,3,N) -> (16,N,3) point transpose
// ---------------------------------------------------------------------------
__global__ void k_transpose(const float* __restrict__ x, float* __restrict__ pts,
                            int N, int total) {
  int i = blockIdx.x * blockDim.x + threadIdx.x;
  if (i >= total) return;
  int b = i / N, n = i % N;
  const float* xb = x + (size_t)b * 3 * N;
  pts[(size_t)i * 3 + 0] = xb[n];
  pts[(size_t)i * 3 + 1] = xb[N + n];
  pts[(size_t)i * 3 + 2] = xb[2 * N + n];
}

// ---------------------------------------------------------------------------
// f32 -> f16 convert with zero padding: out is Rp x Kp, in is R x K.
// Padding rows/cols are zero so the WMMA GEMM needs no K/N guards at all.
// ---------------------------------------------------------------------------
__global__ void k_cvt_pad(const float* __restrict__ in, _Float16* __restrict__ out,
                          int R, int K, int Rp, int Kp) {
  int i = blockIdx.x * blockDim.x + threadIdx.x;
  if (i >= Rp * Kp) return;
  int r = i / Kp, k = i % Kp;
  out[i] = (r < R && k < K) ? (_Float16)in[(size_t)r * K + k] : (_Float16)0.0f;
}

// ---------------------------------------------------------------------------
// WMMA GEMM on pre-converted, zero-padded f16 operands.
//   C[M,N] = A16[M,Kp] * W16[Np,Kp]^T (+bias)
// One wave32 per 16x16 C tile. Inner loop is guard-free:
//   4x global_load_b128 + v_wmma_f32_16x16x32_f16.
// Fragment layouts (ISA 7.12.2):
//   A: lane<16 -> M=lane, halves {k0..k0+7},{k0+16..k0+23}; lane>=16 -> +8.
//   B: lane -> N=lane&15, halves {k0..k0+7},{k0+16..k0+23}; lane>=16 -> +8.
//   C/D: lane -> N=lane&15, vgpr r -> M = r + (lane>=16 ? 8 : 0).
// M must be a multiple of 16 (true for every layer here).
// ---------------------------------------------------------------------------
__global__ void k_gemm_wmma(const _Float16* __restrict__ A16,
                            const _Float16* __restrict__ W16,
                            const float* __restrict__ bias, float* __restrict__ C,
                            int M, int N, int Kp) {
  const int lane = threadIdx.x;          // wave32
  const int n0 = blockIdx.x * 16;
  const int m0 = blockIdx.y * 16;
  const int lc = lane & 15;
  const bool hi = lane >= 16;

  const _Float16* pa = A16 + (size_t)(m0 + lc) * Kp + (hi ? 8 : 0);
  const _Float16* pb = W16 + (size_t)(n0 + lc) * Kp + (hi ? 8 : 0);

  v8f acc = {};
  for (int k0 = 0; k0 < Kp; k0 += 32) {
    v8h a0 = *(const v8h*)(pa);
    v8h a1 = *(const v8h*)(pa + 16);
    v8h b0 = *(const v8h*)(pb);
    v8h b1 = *(const v8h*)(pb + 16);
    v16h a = __builtin_shufflevector(a0, a1, 0, 1, 2, 3, 4, 5, 6, 7,
                                     8, 9, 10, 11, 12, 13, 14, 15);
    v16h b = __builtin_shufflevector(b0, b1, 0, 1, 2, 3, 4, 5, 6, 7,
                                     8, 9, 10, 11, 12, 13, 14, 15);
    acc = __builtin_amdgcn_wmma_f32_16x16x32_f16(false, a, false, b,
                                                 (short)0, acc, false, false);
    pa += 32;
    pb += 32;
  }

  int col = n0 + lc;
  if (col < N) {
    float bv = bias ? bias[col] : 0.0f;
#pragma unroll
    for (int r = 0; r < 8; ++r) {
      int row = m0 + r + (hi ? 8 : 0);
      C[(size_t)row * N + col] = acc[r] + bv;
    }
  }
}

// ---------------------------------------------------------------------------
// Per-column mean/var over R rows (BatchNorm stats, population variance)
// ---------------------------------------------------------------------------
__global__ void k_colstats(const float* __restrict__ Y, int R, int C,
                           float* __restrict__ mv) {
  int c = blockIdx.x;
  float s = 0.f, ss = 0.f;
  for (int r = threadIdx.x; r < R; r += blockDim.x) {
    float y = Y[(size_t)r * C + c];
    s += y; ss += y * y;
  }
  __shared__ float s1[256], s2[256];
  s1[threadIdx.x] = s; s2[threadIdx.x] = ss;
  __syncthreads();
  for (int w = blockDim.x >> 1; w > 0; w >>= 1) {
    if (threadIdx.x < w) {
      s1[threadIdx.x] += s1[threadIdx.x + w];
      s2[threadIdx.x] += s2[threadIdx.x + w];
    }
    __syncthreads();
  }
  if (threadIdx.x == 0) {
    float m = s1[0] / (float)R;
    mv[c] = m;
    mv[C + c] = s2[0] / (float)R - m * m;
  }
}

// ---------------------------------------------------------------------------
// In-place BatchNorm + LeakyReLU(0.2)
// ---------------------------------------------------------------------------
__global__ void k_bn_lrelu(float* __restrict__ Y, int total, int C,
                           const float* __restrict__ mv,
                           const float* __restrict__ g,
                           const float* __restrict__ b) {
  int i = blockIdx.x * blockDim.x + threadIdx.x;
  if (i >= total) return;
  int c = i % C;
  float y = (Y[i] - mv[c]) * rsqrtf(mv[C + c] + BN_EPS) * g[c] + b[c];
  Y[i] = y > 0.f ? y : LRELU * y;
}

// ---------------------------------------------------------------------------
// Fused KNN: each query scans candidates tiled through LDS, maintains its
// 20 nearest in sorted registers. The B x M x N distance matrix (256 MB for
// the first pool) is never materialized.
// ---------------------------------------------------------------------------
__global__ void k_knn(const float* __restrict__ pts, int batchStride,
                      int nPts, int nQ, int* __restrict__ outIdx) {
  const int b = blockIdx.y;
  const int q = blockIdx.x * blockDim.x + threadIdx.x;
  const float* P = pts + (size_t)b * batchStride;
  const bool act = q < nQ;
  float qx = 0.f, qy = 0.f, qz = 0.f;
  if (act) { qx = P[q * 3]; qy = P[q * 3 + 1]; qz = P[q * 3 + 2]; }

  float bd[KNN]; int bi[KNN];
#pragma unroll
  for (int k = 0; k < KNN; ++k) { bd[k] = 3.0e38f; bi[k] = 0; }

  __shared__ float sx[128], sy[128], sz[128];
  for (int n0 = 0; n0 < nPts; n0 += 128) {
    __syncthreads();
    int n = n0 + threadIdx.x;
    if (n < nPts) {
      sx[threadIdx.x] = P[n * 3];
      sy[threadIdx.x] = P[n * 3 + 1];
      sz[threadIdx.x] = P[n * 3 + 2];
    }
    if (n0 + 128 < nPts)  // hint next tile toward L2 (global_prefetch)
      __builtin_prefetch(&P[(n0 + 128 + threadIdx.x) * 3], 0, 1);
    __syncthreads();
    int cnt = nPts - n0; if (cnt > 128) cnt = 128;
    if (act) {
      for (int j = 0; j < cnt; ++j) {
        float dx = sx[j] - qx, dy = sy[j] - qy, dz = sz[j] - qz;
        float d2 = dx * dx + dy * dy + dz * dz;
        if (d2 < bd[KNN - 1]) {           // sorted insert, fully unrolled
          float v = d2; int vi = n0 + j;
#pragma unroll
          for (int k = 0; k < KNN; ++k) {
            if (v < bd[k]) {
              float tv = bd[k]; bd[k] = v; v = tv;
              int   ti = bi[k]; bi[k] = vi; vi = ti;
            }
          }
        }
      }
    }
  }
  if (act) {
    int* o = outIdx + ((size_t)b * nQ + q) * KNN;
#pragma unroll
    for (int k = 0; k < KNN; ++k) o[k] = bi[k];
  }
}

// ---------------------------------------------------------------------------
// Gather + channel max over the 20 neighbors; writes into a channel-offset
// slice of the output (implements the concat for free).
// ---------------------------------------------------------------------------
__global__ void k_gather_max(const float* __restrict__ feat, int featBatchStride, int C,
                             const int* __restrict__ idx, int nQ,
                             float* __restrict__ out, int outRowStride, int outColOff,
                             int outBatchStride) {
  const int b = blockIdx.y, q = blockIdx.x;
  __shared__ int si[KNN];
  const int* ip = idx + ((size_t)b * nQ + q) * KNN;
  if (threadIdx.x < KNN) si[threadIdx.x] = ip[threadIdx.x];
  __syncthreads();
  const float* F = feat + (size_t)b * featBatchStride;
  for (int c = threadIdx.x; c < C; c += blockDim.x) {
    float m = -3.0e38f;
#pragma unroll
    for (int k = 0; k < KNN; ++k) m = fmaxf(m, F[(size_t)si[k] * C + c]);
    out[(size_t)b * outBatchStride + (size_t)q * outRowStride + outColOff + c] = m;
  }
}

// ---------------------------------------------------------------------------
// (B,P,C) -> (B,2C): [max over P | mean over P]
// ---------------------------------------------------------------------------
__global__ void k_maxmean(const float* __restrict__ H, int P, int C,
                          float* __restrict__ G, int B) {
  int i = blockIdx.x * blockDim.x + threadIdx.x;
  if (i >= B * C) return;
  int b = i / C, c = i % C;
  const float* p = H + (size_t)b * P * C + c;
  float mx = -3.0e38f, sm = 0.f;
  for (int r = 0; r < P; ++r) { float y = p[(size_t)r * C]; mx = fmaxf(mx, y); sm += y; }
  G[(size_t)b * 2 * C + c] = mx;
  G[(size_t)b * 2 * C + C + c] = sm / (float)P;
}

// ---------------------------------------------------------------------------
extern "C" void kernel_launch(void* const* d_in, const int* in_sizes, int n_in,
                              void* d_out, int out_size, void* d_ws, size_t ws_size,
                              hipStream_t stream) {
  const int B = 16, N = 4096;
  const float* x   = (const float*)d_in[0];
  const float* w1  = (const float*)d_in[1];
  const float* g1  = (const float*)d_in[2];
  const float* b1  = (const float*)d_in[3];
  const float* w2  = (const float*)d_in[4];
  const float* g2  = (const float*)d_in[5];
  const float* b2  = (const float*)d_in[6];
  const float* w3  = (const float*)d_in[7];
  const float* g3  = (const float*)d_in[8];
  const float* b3  = (const float*)d_in[9];
  const float* w4  = (const float*)d_in[10];
  const float* g4  = (const float*)d_in[11];
  const float* b4  = (const float*)d_in[12];
  const float* w5  = (const float*)d_in[13];
  const float* g5  = (const float*)d_in[14];
  const float* b5  = (const float*)d_in[15];
  const float* l1w = (const float*)d_in[16];
  const float* g6  = (const float*)d_in[17];
  const float* b6  = (const float*)d_in[18];
  const float* l2w = (const float*)d_in[19];
  const float* l2b = (const float*)d_in[20];
  const float* g7  = (const float*)d_in[21];
  const float* b7  = (const float*)d_in[22];
  const float* l3w = (const float*)d_in[23];
  const float* l3b = (const float*)d_in[24];
  float* out = (float*)d_out;

  // Bump allocator over workspace (~49 MB total)
  char* ws = (char*)d_ws;
  size_t off = 0;
  auto alloc = [&](size_t bytes) -> void* {
    void* p = ws + off;
    off += (bytes + 255) & ~(size_t)255;
    return p;
  };
  float* pts = (float*)alloc((size_t)B * N * 3 * 4);
  float* F0  = (float*)alloc((size_t)B * N * 64 * 4);       // Y0 / f after bn1
  float* F1p = (float*)alloc((size_t)B * 1024 * 64 * 4);
  float* F2  = (float*)alloc((size_t)B * 1024 * 64 * 4);
  float* F2p = (float*)alloc((size_t)B * 256 * 64 * 4);
  float* F3  = (float*)alloc((size_t)B * 256 * 128 * 4);
  float* F3p = (float*)alloc((size_t)B * 128 * 128 * 4);
  float* F4  = (float*)alloc((size_t)B * 128 * 256 * 4);
  float* H   = (float*)alloc((size_t)B * 64 * 512 * 4);     // concat target
  float* H5  = (float*)alloc((size_t)B * 64 * 1024 * 4);
  float* G   = (float*)alloc((size_t)B * 2048 * 4);
  float* h6  = (float*)alloc((size_t)B * 512 * 4);
  float* h7  = (float*)alloc((size_t)B * 256 * 4);
  float* mv  = (float*)alloc((size_t)2 * 1024 * 4);         // reused BN stats
  _Float16* A16 = (_Float16*)alloc((size_t)B * N * 32 * 2); // 65536 x 32 max
  _Float16* W16 = (_Float16*)alloc((size_t)1024 * 2048 * 2);
  int* idx1 = (int*)alloc((size_t)B * 1024 * KNN * 4);
  int* idx2 = (int*)alloc((size_t)B * 256 * KNN * 4);
  int* idx3 = (int*)alloc((size_t)B * 128 * KNN * 4);
  int* idxA = (int*)alloc((size_t)B * 64 * KNN * 4);
  int* idxB = (int*)alloc((size_t)B * 64 * KNN * 4);
  int* idxC = (int*)alloc((size_t)B * 64 * KNN * 4);

  const int PSTR = N * 3;  // point-set batch stride (all subsets are prefixes)

  auto cvt = [&](const float* in, _Float16* o, int R, int K, int Rp, int Kp) {
    int tot = Rp * Kp;
    k_cvt_pad<<<(tot + 255) / 256, 256, 0, stream>>>(in, o, R, K, Rp, Kp);
  };
  auto gemm = [&](const float* A, const float* W, const float* bias, float* Y,
                  int M, int Nn, int K) {
    int Kp = (K + 31) & ~31;
    int Np = (Nn + 15) & ~15;
    cvt(A, A16, M, K, M, Kp);
    cvt(W, W16, Nn, K, Np, Kp);
    dim3 grid(Np / 16, M / 16);
    k_gemm_wmma<<<grid, 32, 0, stream>>>(A16, W16, bias, Y, M, Nn, Kp);
  };
  auto gemm_bn = [&](const float* A, const float* W, const float* bias, float* Y,
                     int M, int Nn, int K, const float* g, const float* bb) {
    gemm(A, W, bias, Y, M, Nn, K);
    k_colstats<<<Nn, 256, 0, stream>>>(Y, M, Nn, mv);
    int tot = M * Nn;
    k_bn_lrelu<<<(tot + 255) / 256, 256, 0, stream>>>(Y, tot, Nn, mv, g, bb);
  };
  auto knn = [&](int nPts, int nQ, int* idx) {
    dim3 grid((nQ + 127) / 128, B);
    k_knn<<<grid, 128, 0, stream>>>(pts, PSTR, nPts, nQ, idx);
  };
  auto gmax = [&](const float* feat, int fbs, int C, const int* idx, int nQ,
                  float* o, int rs, int coff, int obs) {
    dim3 grid(nQ, B);
    k_gather_max<<<grid, 64, 0, stream>>>(feat, fbs, C, idx, nQ, o, rs, coff, obs);
  };

  // pts = transpose(x); f = bn_lrelu(pts @ w1^T)
  k_transpose<<<(B * N + 255) / 256, 256, 0, stream>>>(x, pts, N, B * N);
  gemm_bn(pts, w1, nullptr, F0, B * N, 64, 3, g1, b1);

  // pool chain (all point subsets are prefixes of pts)
  knn(4096, 1024, idx1);                                   // pool(pts, f, 1024)
  gmax(F0, N * 64, 64, idx1, 1024, F1p, 64, 0, 1024 * 64);
  knn(1024, 64, idxA);                                     // x1 = pool(p1, f1p, 64)
  gmax(F1p, 1024 * 64, 64, idxA, 64, H, 512, 0, 64 * 512);

  gemm_bn(F1p, w2, nullptr, F2, B * 1024, 64, 64, g2, b2); // f = bn(f @ w2^T)
  knn(1024, 256, idx2);                                    // pool(p1, f2, 256)
  gmax(F2, 1024 * 64, 64, idx2, 256, F2p, 64, 0, 256 * 64);
  knn(256, 64, idxB);                                      // x2 = pool(p2, f2p, 64)
  gmax(F2p, 256 * 64, 64, idxB, 64, H, 512, 64, 64 * 512);

  gemm_bn(F2p, w3, nullptr, F3, B * 256, 128, 64, g3, b3); // f = bn(f @ w3^T)
  knn(256, 128, idx3);                                     // pool(p2, f3, 128)
  gmax(F3, 256 * 128, 128, idx3, 128, F3p, 128, 0, 128 * 128);
  knn(128, 64, idxC);            // shared by x3-pool and f4-pool (same queries!)
  gmax(F3p, 128 * 128, 128, idxC, 64, H, 512, 128, 64 * 512);  // x3

  gemm_bn(F3p, w4, nullptr, F4, B * 128, 256, 128, g4, b4);    // f = bn(f @ w4^T)
  gmax(F4, 128 * 256, 256, idxC, 64, H, 512, 256, 64 * 512);   // f4p -> concat

  // h = bn(concat @ w5^T); global max|mean over points; FC head
  gemm_bn(H, w5, nullptr, H5, B * 64, 1024, 512, g5, b5);
  k_maxmean<<<(B * 1024 + 255) / 256, 256, 0, stream>>>(H5, 64, 1024, G, B);
  gemm_bn(G, l1w, nullptr, h6, B, 512, 2048, g6, b6);
  gemm_bn(h6, l2w, l2b, h7, B, 256, 512, g7, b7);
  gemm(h7, l3w, l3b, out, B, 40, 256);
}
